// ConditionalGraphNetwork_7507602833607
// MI455X (gfx1250) — matrix operand order repaired and checked by reading, not compiled, not run on hardware
//
#include <hip/hip_runtime.h>
#include <hip/hip_bf16.h>
#include <stdint.h>

// ---------------------------------------------------------------------------
// CDNA5 (gfx1250) wave32 WMMA bf16 types & helpers
// ---------------------------------------------------------------------------
typedef __attribute__((ext_vector_type(16))) __bf16 v16bf;
typedef __attribute__((ext_vector_type(8)))  float  v8f;

union FragB16 { v16bf v; unsigned u[8]; };

__device__ __forceinline__ unsigned short f2bf(float a) {
  unsigned u = __float_as_uint(a);
  u = (u + 0x7FFFu + ((u >> 16) & 1u)) >> 16;   // round-to-nearest-even
  return (unsigned short)u;
}
__device__ __forceinline__ unsigned f2bf_pair(float lo, float hi) {
  return (unsigned)f2bf(lo) | ((unsigned)f2bf(hi) << 16);
}
__device__ __forceinline__ float bf2f(unsigned short b) {
  return __uint_as_float(((unsigned)b) << 16);
}

// ISA 7.12.2: 16-bit A-matrix 16x32 layout. lane half (lane>>4) shifts K by +8,
// VGPRs 0..3 hold K {0,2,4,6}+, VGPRs 4..7 hold K 16+{0,2,4,6}+.
__device__ __forceinline__ int a_kbase(int v, int half, int kt) {
  return kt * 32 + ((v < 4) ? (v * 2) : (16 + (v - 4) * 2)) + half * 8;
}

#define HDIM 128

// ---------------------------------------------------------------------------
// Weight packer: W[K=128][N=128] row-major f32 -> bf16 B-fragments.
// B 32x16 bf16 layout: lane col = lane&15, K = kt*32 + (lane>>4)*8 + (v<4 ? v*2 : 16+(v-4)*2).
// Packed index: ((c*4 + kt)*32 + lane)*8 + v  -> per-lane contiguous 32B.
// ---------------------------------------------------------------------------
__global__ void k_pack(const float* __restrict__ W, unsigned* __restrict__ out) {
  int idx = blockIdx.x * blockDim.x + threadIdx.x;
  if (idx >= 8192) return;
  int v    = idx & 7;
  int lane = (idx >> 3) & 31;
  int kt   = (idx >> 8) & 3;
  int c    = idx >> 10;
  int col  = c * 16 + (lane & 15);
  int k    = kt * 32 + (lane >> 4) * 8 + ((v < 4) ? (v * 2) : (16 + (v - 4) * 2));
  out[idx] = f2bf_pair(W[k * HDIM + col], W[(k + 1) * HDIM + col]);
}

__global__ void k_zero(float* __restrict__ p, int n) {
  int i = blockIdx.x * blockDim.x + threadIdx.x;
  if (i < n) p[i] = 0.0f;
}

__global__ void k_cnt(const long long* __restrict__ dst, float* __restrict__ cnt, int E) {
  int e = blockIdx.x * blockDim.x + threadIdx.x;
  if (e < E) atomicAdd(&cnt[(int)dst[e]], 1.0f);
}

__global__ void k_inv(float* __restrict__ cnt, int N) {
  int i = blockIdx.x * blockDim.x + threadIdx.x;
  if (i < N) cnt[i] = 1.0f / fmaxf(cnt[i], 1.0f);
}

// time_mlp + condition_mlp, B=4 (tiny; one block of 128 threads)
__global__ void k_timecond(const float* __restrict__ t, const float* __restrict__ cond,
                           const float* __restrict__ tw1, const float* __restrict__ tb1,
                           const float* __restrict__ tw2, const float* __restrict__ tb2,
                           const float* __restrict__ cw1, const float* __restrict__ cb1,
                           const float* __restrict__ cw2, const float* __restrict__ cb2,
                           float* __restrict__ tf, float* __restrict__ cf) {
  __shared__ float sh[4][HDIM];
  int j = threadIdx.x;
  for (int b = 0; b < 4; ++b)
    sh[b][j] = fmaxf(t[b] * tw1[j] + tb1[j], 0.0f);
  __syncthreads();
  for (int b = 0; b < 4; ++b) {
    float s = tb2[j];
    for (int k = 0; k < HDIM; ++k) s += sh[b][k] * tw2[k * HDIM + j];
    tf[b * HDIM + j] = s;
  }
  __syncthreads();
  for (int b = 0; b < 4; ++b) {
    float s = cb1[j];
    for (int d = 0; d < 100; ++d) s += cond[b * 100 + d] * cw1[d * HDIM + j];
    sh[b][j] = fmaxf(s, 0.0f);
  }
  __syncthreads();
  for (int b = 0; b < 4; ++b) {
    float s = cb2[j];
    for (int k = 0; k < HDIM; ++k) s += sh[b][k] * cw2[k * HDIM + j];
    cf[b * HDIM + j] = s;
  }
}

// input_mlp first layer (K=3, VALU) -> bf16 row-major
__global__ void k_hid(const float* __restrict__ x, const float* __restrict__ w1,
                      const float* __restrict__ b1, unsigned short* __restrict__ hidb, int N) {
  int i = blockIdx.x * blockDim.x + threadIdx.x;
  if (i >= N * HDIM) return;
  int n = i >> 7, j = i & 127;
  float s = b1[j] + x[n * 3 + 0] * w1[j] + x[n * 3 + 1] * w1[HDIM + j] + x[n * 3 + 2] * w1[2 * HDIM + j];
  hidb[i] = f2bf(fmaxf(s, 0.0f));
}

// Load A-fragment from a bf16 row-major [*,128] buffer viewed as uint pairs.
__device__ __forceinline__ void load_afrag(const unsigned* __restrict__ rowbase,
                                           int lane, int kt, FragB16& A) {
  int half = lane >> 4;
#pragma unroll
  for (int v = 0; v < 8; ++v) {
    int kb = a_kbase(v, half, kt);
    A.u[v] = rowbase[kb >> 1];
  }
}

// input_mlp second layer + time/cond features: h = hid @ in_w2 + b2 + tf[batch] + cf[batch]
__global__ void k_gemm_in(const unsigned* __restrict__ hidu, const unsigned* __restrict__ w2p,
                          const float* __restrict__ b2, const long long* __restrict__ batch,
                          const float* __restrict__ tf, const float* __restrict__ cf,
                          float* __restrict__ h, unsigned short* __restrict__ hb, int N) {
  int rt = blockIdx.x, tid = threadIdx.x, wave = tid >> 5, lane = tid & 31;
  int row0 = rt * 16 + (lane & 15);
  if (row0 >= N) row0 = N - 1;
  const unsigned* abase = hidu + (size_t)row0 * 64;
  v8f acc = {};
  FragB16 A, B;
#pragma unroll
  for (int kt = 0; kt < 4; ++kt) {
    load_afrag(abase, lane, kt, A);
    const unsigned* bp = w2p + ((size_t)(wave * 4 + kt) * 32 + lane) * 8;
#pragma unroll
    for (int v = 0; v < 8; ++v) B.u[v] = bp[v];
    acc = __builtin_amdgcn_wmma_f32_16x16x32_bf16(false, A.v, false, B.v, (short)0, acc, false, false);
  }
  int half = lane >> 4, colg = wave * 16 + (lane & 15);
  float bb = b2[colg];
#pragma unroll
  for (int v = 0; v < 8; ++v) {
    int row = rt * 16 + v + half * 8;
    if (row < N) {
      int b = (int)batch[row];
      float s = acc[v] + bb + tf[b * HDIM + colg] + cf[b * HDIM + colg];
      h[(size_t)row * HDIM + colg]  = s;
      hb[(size_t)row * HDIM + colg] = f2bf(s);
    }
  }
}

// Per-layer node precompute: Pa = h@W1a + b1, Pb = h@W1b  (shared A-fragments)
__global__ void k_pre(const unsigned* __restrict__ hbu, const unsigned* __restrict__ w1ap,
                      const unsigned* __restrict__ w1bp, const float* __restrict__ b1,
                      float* __restrict__ Pa, float* __restrict__ Pb, int N) {
  int rt = blockIdx.x, tid = threadIdx.x, wave = tid >> 5, lane = tid & 31;
  int row0 = rt * 16 + (lane & 15);
  if (row0 >= N) row0 = N - 1;
  const unsigned* abase = hbu + (size_t)row0 * 64;
  v8f accA = {}, accB = {};
  FragB16 A, Ba, Bb;
#pragma unroll
  for (int kt = 0; kt < 4; ++kt) {
    load_afrag(abase, lane, kt, A);
    const unsigned* bpa = w1ap + ((size_t)(wave * 4 + kt) * 32 + lane) * 8;
    const unsigned* bpb = w1bp + ((size_t)(wave * 4 + kt) * 32 + lane) * 8;
#pragma unroll
    for (int v = 0; v < 8; ++v) { Ba.u[v] = bpa[v]; Bb.u[v] = bpb[v]; }
    accA = __builtin_amdgcn_wmma_f32_16x16x32_bf16(false, A.v, false, Ba.v, (short)0, accA, false, false);
    accB = __builtin_amdgcn_wmma_f32_16x16x32_bf16(false, A.v, false, Bb.v, (short)0, accB, false, false);
  }
  int half = lane >> 4, colg = wave * 16 + (lane & 15);
  float bb = b1[colg];
#pragma unroll
  for (int v = 0; v < 8; ++v) {
    int row = rt * 16 + v + half * 8;
    if (row < N) {
      Pa[(size_t)row * HDIM + colg] = accA[v] + bb;
      Pb[(size_t)row * HDIM + colg] = accB[v];
    }
  }
}

// Edge kernel: pre = relu(Pa[dst]+Pb[src]+e*w1e) tile (16 edges) -> LDS bf16,
// msg = pre @ W2 + b2 via WMMA, scatter-add into agg[dst].
// W2 fragments, b2 and per-thread w1e pairs are register-resident across tiles.
__global__ void k_edge(const float* __restrict__ Pa, const float* __restrict__ Pb,
                       const long long* __restrict__ src, const long long* __restrict__ dst,
                       const float* __restrict__ eattr, const unsigned* __restrict__ w2p,
                       const float* __restrict__ b2, const float* __restrict__ w1e,
                       float* __restrict__ agg, int E) {
  __shared__ unsigned lds_pre[16 * 64];
  __shared__ int   lds_dst[16];
  __shared__ int   lds_src[16];
  __shared__ float lds_ea[16];
  int numTiles = (E + 15) >> 4;
  int tid = threadIdx.x, wave = tid >> 5, lane = tid & 31;
  int half = lane >> 4;

  // ---- loop-invariant hoists ----
  FragB16 Bw[4];
#pragma unroll
  for (int kt = 0; kt < 4; ++kt) {
    const unsigned* bp = w2p + ((size_t)(wave * 4 + kt) * 32 + lane) * 8;
#pragma unroll
    for (int v = 0; v < 8; ++v) Bw[kt].u[v] = bp[v];
  }
  int colg = wave * 16 + (lane & 15);
  float bb = b2[colg];
  float we0[4], we1[4];
#pragma unroll
  for (int i = 0; i < 4; ++i) {
    int cp = (tid + 256 * i) & 63;
    we0[i] = w1e[cp * 2];
    we1[i] = w1e[cp * 2 + 1];
  }

  for (int tile = blockIdx.x; tile < numTiles; tile += gridDim.x) {
    if (tid < 16) {
      int e = tile * 16 + tid;
      if (e < E) { lds_dst[tid] = (int)dst[e]; lds_src[tid] = (int)src[e]; lds_ea[tid] = eattr[e]; }
      else       { lds_dst[tid] = 0;           lds_src[tid] = 0;           lds_ea[tid] = 0.0f; }
    }
    __syncthreads();
    // phase 1: build relu(pre) tile in LDS (bias b1 already folded into Pa)
#pragma unroll
    for (int i = 0; i < 4; ++i) {
      int pi = tid + 256 * i;            // 1024 uint pairs
      int el = pi >> 6, cp = pi & 63;
      bool valid = (tile * 16 + el) < E;
      const float2* pa2 = (const float2*)(Pa + (size_t)lds_dst[el] * HDIM);
      const float2* pb2 = (const float2*)(Pb + (size_t)lds_src[el] * HDIM);
      float2 av = pa2[cp];
      float2 bv = pb2[cp];
      float ea = lds_ea[el];
      float v0 = fmaxf(av.x + bv.x + ea * we0[i], 0.0f);
      float v1 = fmaxf(av.y + bv.y + ea * we1[i], 0.0f);
      if (!valid) { v0 = 0.0f; v1 = 0.0f; }
      lds_pre[el * 64 + cp] = f2bf_pair(v0, v1);
    }
    __syncthreads();
    // phase 2: 8 waves x one 16-col tile; K=128 -> 4 WMMAs, W2 in registers
    v8f acc = {};
    FragB16 A;
#pragma unroll
    for (int kt = 0; kt < 4; ++kt) {
      int row = lane & 15;
#pragma unroll
      for (int v = 0; v < 8; ++v) {
        int kb = a_kbase(v, half, kt);
        A.u[v] = lds_pre[row * 64 + (kb >> 1)];
      }
      acc = __builtin_amdgcn_wmma_f32_16x16x32_bf16(false, A.v, false, Bw[kt].v, (short)0, acc, false, false);
    }
#pragma unroll
    for (int v = 0; v < 8; ++v) {
      int el = v + half * 8;
      int e  = tile * 16 + el;
      if (e < E) atomicAdd(&agg[(size_t)lds_dst[el] * HDIM + colg], acc[v] + bb);
    }
    __syncthreads();
  }
}

// scatter-mean scale + relu + residual; re-zero agg; maintain bf16 mirror
__global__ void k_update(float* __restrict__ agg, const float* __restrict__ inv,
                         float* __restrict__ h, unsigned short* __restrict__ hb, int total) {
  int i = blockIdx.x * blockDim.x + threadIdx.x;
  if (i >= total) return;
  int row = i >> 7;
  float v = fmaxf(agg[i] * inv[row], 0.0f) + h[i];
  h[i]  = v;
  hb[i] = f2bf(v);
  agg[i] = 0.0f;
}

// output_mlp first layer: t1 = relu(h @ out_w1 + out_b1) -> bf16
__global__ void k_out1(const unsigned* __restrict__ hbu, const unsigned* __restrict__ w1p,
                       const float* __restrict__ b1, unsigned short* __restrict__ t1, int N) {
  int rt = blockIdx.x, tid = threadIdx.x, wave = tid >> 5, lane = tid & 31;
  int row0 = rt * 16 + (lane & 15);
  if (row0 >= N) row0 = N - 1;
  const unsigned* abase = hbu + (size_t)row0 * 64;
  v8f acc = {};
  FragB16 A, B;
#pragma unroll
  for (int kt = 0; kt < 4; ++kt) {
    load_afrag(abase, lane, kt, A);
    const unsigned* bp = w1p + ((size_t)(wave * 4 + kt) * 32 + lane) * 8;
#pragma unroll
    for (int v = 0; v < 8; ++v) B.u[v] = bp[v];
    acc = __builtin_amdgcn_wmma_f32_16x16x32_bf16(false, A.v, false, B.v, (short)0, acc, false, false);
  }
  int half = lane >> 4, colg = wave * 16 + (lane & 15);
  float bb = b1[colg];
#pragma unroll
  for (int v = 0; v < 8; ++v) {
    int row = rt * 16 + v + half * 8;
    if (row < N) t1[(size_t)row * HDIM + colg] = f2bf(fmaxf(acc[v] + bb, 0.0f));
  }
}

// output_mlp second layer: [128]->[3] per node (VALU)
__global__ void k_out2(const unsigned short* __restrict__ t1, const float* __restrict__ w2,
                       const float* __restrict__ b2, float* __restrict__ out, int N) {
  int n = blockIdx.x * blockDim.x + threadIdx.x;
  if (n >= N) return;
  float a0 = b2[0], a1 = b2[1], a2 = b2[2];
  const unsigned short* r = t1 + (size_t)n * HDIM;
#pragma unroll 4
  for (int k = 0; k < HDIM; ++k) {
    float a = bf2f(r[k]);
    a0 += a * w2[k * 3 + 0];
    a1 += a * w2[k * 3 + 1];
    a2 += a * w2[k * 3 + 2];
  }
  out[n * 3 + 0] = a0;
  out[n * 3 + 1] = a1;
  out[n * 3 + 2] = a2;
}

// ---------------------------------------------------------------------------
extern "C" void kernel_launch(void* const* d_in, const int* in_sizes, int n_in,
                              void* d_out, int out_size, void* d_ws, size_t ws_size,
                              hipStream_t stream) {
  (void)n_in; (void)out_size; (void)ws_size;
  const float*     x      = (const float*)d_in[0];
  const long long* eidx   = (const long long*)d_in[1];
  const float*     eattr  = (const float*)d_in[2];
  const float*     t      = (const float*)d_in[3];
  const long long* batch  = (const long long*)d_in[4];
  const float*     cond   = (const float*)d_in[5];
  const float *in_w1 = (const float*)d_in[6],  *in_b1 = (const float*)d_in[7];
  const float *in_w2 = (const float*)d_in[8],  *in_b2 = (const float*)d_in[9];
  const float *t_w1  = (const float*)d_in[10], *t_b1  = (const float*)d_in[11];
  const float *t_w2  = (const float*)d_in[12], *t_b2  = (const float*)d_in[13];
  const float *c_w1  = (const float*)d_in[14], *c_b1  = (const float*)d_in[15];
  const float *c_w2  = (const float*)d_in[16], *c_b2  = (const float*)d_in[17];
  const float *convW1 = (const float*)d_in[18], *convb1 = (const float*)d_in[19];
  const float *convW2 = (const float*)d_in[20], *convb2 = (const float*)d_in[21];
  const float *out_w1 = (const float*)d_in[22], *out_b1 = (const float*)d_in[23];
  const float *out_w2 = (const float*)d_in[24], *out_b2 = (const float*)d_in[25];

  const int N = in_sizes[0] / 3;
  const int E = in_sizes[2];
  const int L = in_sizes[19] / HDIM;   // convb1 is [L,128]
  const long long* srcI = eidx;        // edge_index[0]
  const long long* dstI = eidx + E;    // edge_index[1]

  // ---- workspace carve-out ----
  char* ws = (char*)d_ws;
  float*          h    = (float*)ws;          ws += (size_t)N * HDIM * 4;
  unsigned short* hb   = (unsigned short*)ws; ws += (size_t)N * HDIM * 2;
  float*          Pa   = (float*)ws;          ws += (size_t)N * HDIM * 4;
  float*          Pb   = (float*)ws;          ws += (size_t)N * HDIM * 4;
  float*          agg  = (float*)ws;          ws += (size_t)N * HDIM * 4;
  float*          inv  = (float*)ws;          ws += (size_t)N * 4;
  float*          tf   = (float*)ws;          ws += 4 * HDIM * 4;
  float*          cf   = (float*)ws;          ws += 4 * HDIM * 4;
  unsigned short* hidb = (unsigned short*)ws; ws += (size_t)N * HDIM * 2;
  unsigned*       wpk  = (unsigned*)ws;       // 20 * 8192 dwords

  const size_t PK = 8192;
  unsigned* pk_inw2  = wpk;
  unsigned* pk_outw1 = wpk + PK;
  auto pk_w1a = [&](int l) { return wpk + (2 + 3 * (size_t)l) * PK; };
  auto pk_w1b = [&](int l) { return wpk + (3 + 3 * (size_t)l) * PK; };
  auto pk_w2  = [&](int l) { return wpk + (4 + 3 * (size_t)l) * PK; };

  const int rowTiles = (N + 15) / 16;
  const int TOT = N * HDIM;

  // ---- setup: zero, pack weights, degree counts, tiny MLPs ----
  k_zero<<<(TOT + 255) / 256, 256, 0, stream>>>(agg, TOT);
  k_zero<<<(N + 255) / 256, 256, 0, stream>>>(inv, N);

  k_pack<<<32, 256, 0, stream>>>(in_w2, pk_inw2);
  k_pack<<<32, 256, 0, stream>>>(out_w1, pk_outw1);
  for (int l = 0; l < L; ++l) {
    const float* W1base = convW1 + (size_t)l * 257 * HDIM;
    k_pack<<<32, 256, 0, stream>>>(W1base, pk_w1a(l));                 // rows 0..127
    k_pack<<<32, 256, 0, stream>>>(W1base + 128 * HDIM, pk_w1b(l));    // rows 128..255
    k_pack<<<32, 256, 0, stream>>>(convW2 + (size_t)l * HDIM * HDIM, pk_w2(l));
  }

  k_cnt<<<(E + 255) / 256, 256, 0, stream>>>(dstI, inv, E);
  k_inv<<<(N + 255) / 256, 256, 0, stream>>>(inv, N);

  k_timecond<<<1, HDIM, 0, stream>>>(t, cond, t_w1, t_b1, t_w2, t_b2,
                                     c_w1, c_b1, c_w2, c_b2, tf, cf);

  // ---- node embedding ----
  k_hid<<<(TOT + 255) / 256, 256, 0, stream>>>(x, in_w1, in_b1, hidb, N);
  k_gemm_in<<<rowTiles, 256, 0, stream>>>((const unsigned*)hidb, pk_inw2, in_b2,
                                          batch, tf, cf, h, hb, N);

  // ---- L EdgeConv layers ----
  for (int l = 0; l < L; ++l) {
    const float* w1e = convW1 + ((size_t)l * 257 + 256) * HDIM;  // edge_attr row of W1
    k_pre<<<rowTiles, 256, 0, stream>>>((const unsigned*)hb, pk_w1a(l), pk_w1b(l),
                                        convb1 + (size_t)l * HDIM, Pa, Pb, N);
    k_edge<<<2048, 256, 0, stream>>>(Pa, Pb, srcI, dstI, eattr, pk_w2(l),
                                     convb2 + (size_t)l * HDIM, w1e, agg, E);
    k_update<<<(TOT + 255) / 256, 256, 0, stream>>>(agg, inv, h, hb, TOT);
  }

  // ---- output MLP ----
  k_out1<<<rowTiles, 256, 0, stream>>>((const unsigned*)hb, pk_outw1, out_b1, hidb, N);
  k_out2<<<(N + 255) / 256, 256, 0, stream>>>(hidb, out_w2, out_b2, (float*)d_out, N);
}
